// CrossVariableAttention_68616397521247
// MI455X (gfx1250) — matrix operand order repaired and checked by reading, not compiled, not run on hardware
//
#include <hip/hip_runtime.h>
#include <hip/hip_bf16.h>

// ---------------------------------------------------------------------------
// CrossVariableAttention fused kernel for gfx1250 (MI455X, wave32, WMMA, TDM)
//
// Shapes (fixed by reference): B=4 L=8 C=256 S=64 W=64, H=4, D=64
// N = B*L*S*W = 131072 rows, channel stride in memory = S*W = 4096 floats.
//
// Plan:
//   prologue kernel : f32 -> f16 weight conversion into d_ws (w_qkv | w_proj)
//   fused kernel    : per block = 64 rows
//     0. TDM tensor_load_to_lds: 64x256 f32 x-tile (strided 4096) -> LDS (1x HBM read)
//     1. LayerNorm stats (fp32) over C=256 per row (from LDS)
//     2. normalize -> f16 tile in LDS (row-major, padded)
//     3. GEMM1 via v_wmma_f32_16x16x32_f16: (768x256) x (256x64) -> qkv LDS f16
//     4. tiny 4x4 cross-head softmax attention per row (fp32 VALU)
//     5. GEMM2 via WMMA: (256x256) x (256x64), epilogue: +shortcut(LDS) +bias, store
// ---------------------------------------------------------------------------

typedef __attribute__((ext_vector_type(16))) _Float16 v16h;
typedef __attribute__((ext_vector_type(8)))  _Float16 v8h;
typedef __attribute__((ext_vector_type(8)))  float    v8f;
typedef __attribute__((ext_vector_type(4)))  unsigned int v4ui;
typedef __attribute__((ext_vector_type(8)))  int          v8si;
typedef __attribute__((ext_vector_type(4)))  int          v4si;

union F16x16 { v16h v; v8h h[2]; _Float16 e[16]; };
union F16x8  { v8h  v; _Float16 e[8]; };
union F32x8  { v8f  v; float e[8]; };

namespace cva {
constexpr int C       = 256;
constexpr int C3      = 768;
constexpr int NHEAD   = 4;
constexpr int DHEAD   = 64;
constexpr int SW      = 4096;     // S*W, channel stride in elements
constexpr int TM      = 64;       // rows per block
constexpr int NROWS   = 131072;   // B*L*S*W
constexpr int THREADS = 256;      // 8 wave32
constexpr float LN_EPS = 1e-5f;
constexpr float SCALE  = 0.125f;  // D^-0.5

constexpr int XN_STRIDE  = C  + 8;   // f16 elements, pad vs bank conflicts
constexpr int QKV_STRIDE = C3 + 8;
constexpr int XAT_STRIDE = C  + 8;

constexpr size_t XS_BYTES   = (size_t)C * TM * 4;      // f32 x tile, [c][r], 64 KB
constexpr size_t XN_BYTES   = (size_t)TM * XN_STRIDE  * 2;
constexpr size_t QKV_BYTES  = (size_t)TM * QKV_STRIDE * 2;
constexpr size_t XAT_BYTES  = (size_t)TM * XAT_STRIDE * 2;
constexpr size_t RED_BYTES  = (size_t)TM * 4 * 2 * 4;  // per-row 4 partial (sum,sumsq)
constexpr size_t RED2_BYTES = (size_t)TM * 2 * 4;      // per-row (mean, rstd)
constexpr size_t SMEM_BYTES = XS_BYTES + XN_BYTES + QKV_BYTES + XAT_BYTES +
                              RED_BYTES + RED2_BYTES;  // ~235 KB (< 320 KB/WGP)

constexpr int WQKV_ELEMS  = C3 * C;  // 196608
constexpr int WPROJ_ELEMS = C * C;   // 65536
constexpr int WTOT = WQKV_ELEMS + WPROJ_ELEMS;
} // namespace cva

// --------------------------- weight f32 -> f16 -----------------------------
__global__ void cva_cvt_weights(const float* __restrict__ wqkv,
                                const float* __restrict__ wproj,
                                _Float16* __restrict__ w16) {
  int i = blockIdx.x * blockDim.x + threadIdx.x;
  if (i < cva::WQKV_ELEMS) {
    w16[i] = (_Float16)wqkv[i];
  } else if (i < cva::WTOT) {
    w16[i] = (_Float16)wproj[i - cva::WQKV_ELEMS];
  }
}

// ------------------------------ fused kernel -------------------------------
__global__ __launch_bounds__(cva::THREADS)
void cva_fused_kernel(const float* __restrict__ x,
                      const _Float16* __restrict__ wq16,   // [768][256] f16
                      const _Float16* __restrict__ wp16,   // [256][256] f16
                      const float* __restrict__ bproj,
                      const float* __restrict__ gamma,
                      const float* __restrict__ beta,
                      float* __restrict__ out) {
  using namespace cva;
  extern __shared__ char smem[];
  float*    XS  = (float*)   (smem);                         // [c][r] f32 tile
  _Float16* XN  = (_Float16*)(smem + XS_BYTES);
  _Float16* QKV = (_Float16*)(smem + XS_BYTES + XN_BYTES);
  _Float16* XAT = (_Float16*)(smem + XS_BYTES + XN_BYTES + QKV_BYTES);
  float*    RED = (float*)   (smem + XS_BYTES + XN_BYTES + QKV_BYTES + XAT_BYTES);
  float*    RED2= (float*)   (smem + XS_BYTES + XN_BYTES + QKV_BYTES + XAT_BYTES + RED_BYTES);

  const int t    = threadIdx.x;
  const int wave = t >> 5;
  const int lane = t & 31;
  const int m16  = lane & 15;
  const int half = lane >> 4;

  const int row0 = blockIdx.x * TM;          // first global row of this tile
  const int bl   = row0 / SW;                // (b*L + l) index
  const int sw0  = row0 % SW;                // offset inside the SW plane
  const float* xblk = x + (size_t)bl * C * SW + sw0;  // + c*SW + r

  // ---- Phase 0: TDM async tensor load of the x tile into LDS ----
  // 2D D#: tile_dim0 = 64 contiguous f32 (rows of one channel),
  //        tile_dim1 = 256 channels, tensor_dim0_stride = 4096 f32.
  // LDS result layout: XS[c*64 + r].
  if (wave == 0) {
    const unsigned long long ga = (unsigned long long)(uintptr_t)xblk;
    const unsigned int lds_addr = (unsigned int)(uintptr_t)XS;  // LDS byte offset

    v4ui g0;
    g0.x = 1u;                                        // count=1, user descriptor
    g0.y = lds_addr;                                  // lds_addr[31:0]
    g0.z = (unsigned int)(ga & 0xffffffffu);          // global_addr[31:0]
    g0.w = (unsigned int)(ga >> 32) | (2u << 30);     // global_addr[56:32] | type=2

    v8si g1;
    g1[0] = (int)(2u << 16);          // wg_mask=0, data_size=2 (4 bytes)
    g1[1] = (int)(64u << 16);         // tensor_dim0 = 64   (bits 79:48, low half)
    g1[2] = (int)(256u << 16);        // tensor_dim1 = 256  (bits 111:80, low half)
    g1[3] = (int)(64u << 16);         // tile_dim0 = 64     (bits 127:112)
    g1[4] = (int)(256u);              // tile_dim1 = 256    (bits 143:128)
    g1[5] = (int)(4096u);             // tensor_dim0_stride = 4096 (bits 207:160)
    g1[6] = 0;
    g1[7] = 0;

    v4si g2 = {0, 0, 0, 0};           // dims 2..4 unused (2D tensor)
    v4si g3 = {0, 0, 0, 0};
    v8si g4 = {0, 0, 0, 0, 0, 0, 0, 0};  // extra group in 6-arg builtin form

    __builtin_amdgcn_tensor_load_to_lds(g0, g1, g2, g3, g4, /*cpol=*/0);
    __builtin_amdgcn_s_wait_tensorcnt(0);
  }
  __syncthreads();

  // ---- Phase 1: LayerNorm statistics (fp32) from LDS ----
  {
    const int r = t & 63;          // row within tile
    const int g = t >> 6;          // channel quarter 0..3
    float s = 0.f, s2 = 0.f;
    #pragma unroll 4
    for (int i = 0; i < 64; ++i) {
      const int c = g * 64 + i;
      float v = XS[c * TM + r];
      s += v; s2 += v * v;
    }
    RED[(r * 4 + g) * 2 + 0] = s;
    RED[(r * 4 + g) * 2 + 1] = s2;
  }
  __syncthreads();
  if (t < TM) {
    float s = 0.f, s2 = 0.f;
    #pragma unroll
    for (int g = 0; g < 4; ++g) {
      s  += RED[(t * 4 + g) * 2 + 0];
      s2 += RED[(t * 4 + g) * 2 + 1];
    }
    float mean = s * (1.0f / C);
    float var  = s2 * (1.0f / C) - mean * mean;
    RED2[t * 2 + 0] = mean;
    RED2[t * 2 + 1] = __frsqrt_rn(var + LN_EPS);
  }
  __syncthreads();

  // ---- Phase 2: normalize, scale by gamma/beta, cast to f16 into LDS ----
  {
    const int r = t & 63;
    const int g = t >> 6;
    const float mean = RED2[r * 2 + 0];
    const float rstd = RED2[r * 2 + 1];
    #pragma unroll 4
    for (int i = 0; i < 64; ++i) {
      const int c = g * 64 + i;
      float v = XS[c * TM + r];
      float xn = (v - mean) * rstd * gamma[c] + beta[c];
      XN[r * XN_STRIDE + c] = (_Float16)xn;
    }
  }
  __syncthreads();

  // ---- Phase 3: GEMM1  qkv[j, row] = sum_c wqkv[j][c] * xn[row][c] ----
  // M = j (768 -> 48 tiles, 6 per wave), N = row (64 -> 4 tiles), K = 256 (8 steps)
  for (int mi = 0; mi < 6; ++mi) {
    const int jbase = (wave * 6 + mi) * 16;
    F32x8 acc[4];
    #pragma unroll
    for (int nt = 0; nt < 4; ++nt) acc[nt].v = (v8f){0,0,0,0,0,0,0,0};

    for (int ks = 0; ks < 8; ++ks) {
      const int kb = ks * 32;
      // A fragment: 16x32 f16.  lane(m16,half): K chunks [kb+8h..+7] and [kb+16+8h..+7]
      const _Float16* arow = wq16 + (size_t)(jbase + m16) * C + kb + 8 * half;
      F16x16 a;
      a.h[0] = *(const v8h*)(arow);
      a.h[1] = *(const v8h*)(arow + 16);
      __builtin_prefetch(arow + C, 0, 3);   // next j-row, keep hot in near cache
      #pragma unroll
      for (int nt = 0; nt < 4; ++nt) {
        // B fragment: 32x16 f16. lane(n=m16, half): K chunks [kb+8h..+7] and [kb+16+8h..+7]
        const _Float16* bp = XN + (nt * 16 + m16) * XN_STRIDE + kb + 8 * half;
        F16x16 b;
        b.h[0] = *(const v8h*)(bp);
        b.h[1] = *(const v8h*)(bp + 16);
        acc[nt].v = __builtin_amdgcn_wmma_f32_16x16x32_f16(
            false, a.v, false, b.v, (short)0, acc[nt].v, false, false);
      }
    }
    // D layout: lane holds row = nt*16 + m16; j = jbase + v + 8*half (v=0..7 contiguous)
    #pragma unroll
    for (int nt = 0; nt < 4; ++nt) {
      const int row = nt * 16 + m16;
      F16x8 pk;
      #pragma unroll
      for (int v = 0; v < 8; ++v) pk.e[v] = (_Float16)acc[nt].e[v];
      *(v8h*)(QKV + row * QKV_STRIDE + jbase + 8 * half) = pk.v;
    }
  }
  __syncthreads();

  // ---- Phase 4: tiny cross-head attention per row (fp32 VALU) ----
  // qkv channel j = d*H*3 + h*3 + t  (t: 0=q 1=k 2=v)
  {
    const int r = t >> 2;           // 64 rows
    const int h = t & 3;            // this thread's query head
    const _Float16* qr = QKV + r * QKV_STRIDE;
    float sc[NHEAD];
    #pragma unroll
    for (int g = 0; g < NHEAD; ++g) {
      float s = 0.f;
      for (int d = 0; d < DHEAD; ++d) {
        float qv = (float)qr[d * 12 + h * 3 + 0];
        float kv = (float)qr[d * 12 + g * 3 + 1];
        s += qv * kv;
      }
      sc[g] = s * SCALE;
    }
    float mx = sc[0];
    #pragma unroll
    for (int g = 1; g < NHEAD; ++g) mx = fmaxf(mx, sc[g]);
    float p[NHEAD], denom = 0.f;
    #pragma unroll
    for (int g = 0; g < NHEAD; ++g) { p[g] = __expf(sc[g] - mx); denom += p[g]; }
    float inv = 1.0f / denom;
    #pragma unroll
    for (int g = 0; g < NHEAD; ++g) p[g] *= inv;
    for (int d = 0; d < DHEAD; ++d) {
      float o = 0.f;
      #pragma unroll
      for (int g = 0; g < NHEAD; ++g)
        o += p[g] * (float)qr[d * 12 + g * 3 + 2];
      // x_attn flattened as (d, h) -> channel d*H + h
      XAT[r * XAT_STRIDE + d * NHEAD + h] = (_Float16)o;
    }
  }
  __syncthreads();

  // ---- Phase 5: GEMM2  proj[i, row] = sum_c wproj[i][c] * xat[row][c] ----
  // M = i (256 -> 16 tiles, 2 per wave), N = row (4 tiles), K = 256 (8 steps)
  for (int mi = 0; mi < 2; ++mi) {
    const int ibase = (wave * 2 + mi) * 16;
    F32x8 acc[4];
    #pragma unroll
    for (int nt = 0; nt < 4; ++nt) acc[nt].v = (v8f){0,0,0,0,0,0,0,0};

    for (int ks = 0; ks < 8; ++ks) {
      const int kb = ks * 32;
      const _Float16* arow = wp16 + (size_t)(ibase + m16) * C + kb + 8 * half;
      F16x16 a;
      a.h[0] = *(const v8h*)(arow);
      a.h[1] = *(const v8h*)(arow + 16);
      #pragma unroll
      for (int nt = 0; nt < 4; ++nt) {
        const _Float16* bp = XAT + (nt * 16 + m16) * XAT_STRIDE + kb + 8 * half;
        F16x16 b;
        b.h[0] = *(const v8h*)(bp);
        b.h[1] = *(const v8h*)(bp + 16);
        acc[nt].v = __builtin_amdgcn_wmma_f32_16x16x32_f16(
            false, a.v, false, b.v, (short)0, acc[nt].v, false, false);
      }
    }
    // Epilogue: out = shortcut(LDS) + proj + bias, stored back to (B,L,C,S,W).
    #pragma unroll
    for (int nt = 0; nt < 4; ++nt) {
      const int rowl = nt * 16 + m16;         // lanes 0..15 -> consecutive rows
      #pragma unroll
      for (int v = 0; v < 8; ++v) {
        const int i = ibase + v + 8 * half;
        const size_t off = (size_t)(bl * C + i) * SW + sw0 + rowl;
        out[off] = XS[i * TM + rowl] + acc[nt].e[v] + bproj[i];
      }
    }
  }
}

// ------------------------------- launcher ----------------------------------
extern "C" void kernel_launch(void* const* d_in, const int* in_sizes, int n_in,
                              void* d_out, int out_size, void* d_ws, size_t ws_size,
                              hipStream_t stream) {
  using namespace cva;
  const float* x      = (const float*)d_in[0];
  const float* w_qkv  = (const float*)d_in[1];
  const float* w_proj = (const float*)d_in[2];
  const float* b_proj = (const float*)d_in[3];
  const float* gamma  = (const float*)d_in[4];
  const float* beta   = (const float*)d_in[5];
  float* out = (float*)d_out;

  _Float16* w16 = (_Float16*)d_ws;   // [WQKV_ELEMS] qkv | [WPROJ_ELEMS] proj

  // 1) convert weights to f16 once (stream-ordered, graph-capture safe)
  {
    int threads = 256;
    int blocks = (WTOT + threads - 1) / threads;
    cva_cvt_weights<<<blocks, threads, 0, stream>>>(w_qkv, w_proj, w16);
  }

  // 2) fused TDM + LN + qkv GEMM + attention + proj GEMM + residual
  //    (idempotent attribute set each call: no static state, capture-safe)
  (void)hipFuncSetAttribute((const void*)cva_fused_kernel,
                            hipFuncAttributeMaxDynamicSharedMemorySize,
                            (int)SMEM_BYTES);
  cva_fused_kernel<<<NROWS / TM, THREADS, SMEM_BYTES, stream>>>(
      x, w16, w16 + WQKV_ELEMS, b_proj, gamma, beta, out);
}
